// StandardAttention_88433376625317
// MI455X (gfx1250) — compile-verified, matrix operand order checked by reading, not run on hardware
//
#include <hip/hip_runtime.h>
#include <hip/hip_bf16.h>

// ---------------------------------------------------------------------------
// Problem constants (match reference)
// ---------------------------------------------------------------------------
#define BATCH 2
#define SEQ   2048
#define HID   2048
#define NH    16
#define NKV   4
#define HD    128
#define GROUPS (NH / NKV)
#define SCALE 0.08838834764831845f   // 128^-0.5

typedef __bf16 v16bf __attribute__((ext_vector_type(16)));
typedef float  v8f   __attribute__((ext_vector_type(8)));
typedef unsigned int u32x4 __attribute__((ext_vector_type(4)));
typedef int          i32x8 __attribute__((ext_vector_type(8)));
typedef int          i32x4 __attribute__((ext_vector_type(4)));

union FragU {
    uint4 u[2];
    v16bf v;
};

__device__ __forceinline__ unsigned short f2bf(float f) {
    unsigned int u = __float_as_uint(f);
    unsigned int r = u + 0x7FFFu + ((u >> 16) & 1u);   // round-to-nearest-even
    return (unsigned short)(r >> 16);
}

// ---------------------------------------------------------------------------
// TDM: issue a 2-D bf16 tile DMA (tile_dim0=32 k-elems, tile_dim1=128 rows)
// from global to LDS, with HW padding 64B-row -> 80B LDS stride (LDSP=40).
// D# layout per CDNA5 ISA sec 8.3/8.4. Tracked by TENSORcnt.
// Toolchain here is the 6-arg builtin flavor (clang-23 / therock-10.0).
// ---------------------------------------------------------------------------
__device__ __forceinline__ void tdm_load_tile_bf16(
    unsigned lds_byte_addr, const unsigned short* gptr,
    unsigned k_total, unsigned rows_total, unsigned long long ld_elems)
{
    unsigned long long ga = (unsigned long long)(uintptr_t)gptr;
    u32x4 g0;
    g0[0] = 1u;                                   // count=1 (valid descriptor)
    g0[1] = lds_byte_addr;                        // lds_addr (bytes)
    g0[2] = (unsigned)ga;                         // global_addr[31:0]
    g0[3] = (unsigned)((ga >> 32) & 0x01FFFFFFu)  // global_addr[56:32]
            | 0x80000000u;                        // type=2 ("image") in [127:126]
    i32x8 g1;
    g1[0] = (int)((1u << 16)        // data_size = 1 -> 2 bytes (bf16)
            | (1u << 20)            // pad_enable
            | (3u << 22)            // pad_interval: 16 DWORDs (one 64B tile row)
            | (3u << 25));          // pad_amount: 4 DWORDs (16B) -> 80B stride
    g1[1] = (int)((k_total & 0xFFFFu) << 16);                 // tensor_dim0[15:0]
    g1[2] = (int)(((k_total >> 16) & 0xFFFFu)                 // tensor_dim0[31:16]
            | ((rows_total & 0xFFFFu) << 16));                // tensor_dim1[15:0]
    g1[3] = (int)(((rows_total >> 16) & 0xFFFFu)              // tensor_dim1[31:16]
            | (32u << 16));                                   // tile_dim0 = 32
    g1[4] = 128;                                              // tile_dim1 = 128
    g1[5] = (int)(unsigned)(ld_elems & 0xFFFFFFFFull);        // dim0_stride[31:0]
    g1[6] = (int)(unsigned)((ld_elems >> 32) & 0xFFFFull);    // dim0_stride[47:32]
    g1[7] = 0;
    i32x4 z4 = {0, 0, 0, 0};                 // group 2 unused (2-D tensor)
    i32x8 z8 = {0, 0, 0, 0, 0, 0, 0, 0};     // group 3 / extra group unused
    __builtin_amdgcn_tensor_load_to_lds(g0, g1, z4, z4, z8, 0);
}

// ---------------------------------------------------------------------------
// Generic WMMA GEMM:  C = alpha * A * B   (bf16 WMMA, f32 accumulate)
//   A: M x K row-major (fp32 or bf16)
//   B: K x N row-major (BTRANS=false)  or  N x K row-major (BTRANS=true)
//   C: row-major (fp32 or bf16)
// Batched over blockIdx.z = zb*nH + zh; B offset uses zh/bHDiv (GQA).
// Block = 256 threads = 8 wave32 waves; tile 128x128x32; each wave owns a
// 32x64 sub-tile (2x4 WMMA accumulators of 16x16).
// When both operands are bf16 and BTRANS (the QK^T case), operand staging is
// done by the Tensor Data Mover with double-buffered LDS; otherwise staging
// is manual with on-the-fly fp32->bf16 conversion.
// ---------------------------------------------------------------------------
#define BM 128
#define BN 128
#define BK 32
#define LDSP 40   // padded LDS row stride (ushorts): 80B keeps b128 alignment

template<bool AF32, bool BF32, bool BTRANS, bool CBF16>
__global__ __launch_bounds__(256) void gemm_wmma_bf16(
    const void* __restrict__ Ap, long long lda,
    const void* __restrict__ Bp, long long ldb,
    void*       __restrict__ Cp, long long ldc,
    int K, float alpha,
    long long aOffB, long long aOffH,
    long long bOffB, long long bOffH, int bHDiv,
    long long cOffB, long long cOffH, int nH)
{
    constexpr bool TDM  = (!AF32 && !BF32 && BTRANS);
    constexpr int  NBUF = TDM ? 2 : 1;
    constexpr int  TILE = BM * LDSP;      // ushorts per tile buffer

    __shared__ alignas(16) unsigned short As[NBUF * TILE];
    __shared__ alignas(16) unsigned short Bs[NBUF * TILE];

    const int zb = blockIdx.z / nH;
    const int zh = blockIdx.z % nH;
    const size_t aBase = (size_t)zb * aOffB + (size_t)zh * aOffH;
    const size_t bBase = (size_t)zb * bOffB + (size_t)(zh / bHDiv) * bOffH;
    const size_t cBase = (size_t)zb * cOffB + (size_t)zh * cOffH;

    const int m0 = blockIdx.y * BM;
    const int n0 = blockIdx.x * BN;

    const int tid  = threadIdx.x;
    const int lane = tid & 31;
    const int wave = tid >> 5;     // 0..7
    const int wm   = wave & 3;     // 4 waves along M
    const int wn   = wave >> 2;    // 2 waves along N
    const int half = lane >> 4;    // hi/lo 16 lanes
    const int l16  = lane & 15;

    v8f acc[2][4];
#pragma unroll
    for (int mi = 0; mi < 2; ++mi)
#pragma unroll
        for (int ni = 0; ni < 4; ++ni)
            acc[mi][ni] = (v8f){0.f, 0.f, 0.f, 0.f, 0.f, 0.f, 0.f, 0.f};

    const int nk = K / BK;

    if constexpr (TDM) {
        // TDM needs the wave-relative LDS byte offsets of the staging buffers.
        const unsigned asOff = (unsigned)(uintptr_t)&As[0];  // low 32 bits = LDS offset
        const unsigned bsOff = (unsigned)(uintptr_t)&Bs[0];
        const unsigned short* aG = (const unsigned short*)Ap + aBase + (size_t)m0 * lda;
        const unsigned short* bG = (const unsigned short*)Bp + bBase + (size_t)n0 * ldb;

        if (wave == 0) {
            tdm_load_tile_bf16(asOff, aG, (unsigned)K, BM, (unsigned long long)lda);
            tdm_load_tile_bf16(bsOff, bG, (unsigned)K, BN, (unsigned long long)ldb);
        }
        for (int it = 0; it < nk; ++it) {
            if (wave == 0) {
                if (it + 1 < nk) {
                    const unsigned bufo = ((it + 1) & 1) * (TILE * 2);  // bytes
                    tdm_load_tile_bf16(asOff + bufo, aG + (it + 1) * BK,
                                       (unsigned)K, BM, (unsigned long long)lda);
                    tdm_load_tile_bf16(bsOff + bufo, bG + (it + 1) * BK,
                                       (unsigned)K, BN, (unsigned long long)ldb);
                    __builtin_amdgcn_s_wait_tensorcnt(2);   // current pair landed
                } else {
                    __builtin_amdgcn_s_wait_tensorcnt(0);
                }
            }
            __syncthreads();

            const unsigned short* AsP = As + (it & 1) * TILE;
            const unsigned short* BsP = Bs + (it & 1) * TILE;
            FragU aF[2], bF[4];
#pragma unroll
            for (int mi = 0; mi < 2; ++mi) {
                const int m = wm * 32 + mi * 16 + l16;
                aF[mi].u[0] = *(const uint4*)&AsP[m * LDSP + half * 8];
                aF[mi].u[1] = *(const uint4*)&AsP[m * LDSP + 16 + half * 8];
            }
#pragma unroll
            for (int ni = 0; ni < 4; ++ni) {
                const int n = wn * 64 + ni * 16 + l16;
                bF[ni].u[0] = *(const uint4*)&BsP[n * LDSP + half * 8];
                bF[ni].u[1] = *(const uint4*)&BsP[n * LDSP + 16 + half * 8];
            }
#pragma unroll
            for (int mi = 0; mi < 2; ++mi)
#pragma unroll
                for (int ni = 0; ni < 4; ++ni)
                    acc[mi][ni] = __builtin_amdgcn_wmma_f32_16x16x32_bf16(
                        false, aF[mi].v, false, bF[ni].v,
                        (short)0, acc[mi][ni], false, false);
            __syncthreads();   // before DMA overwrites this buffer
        }
    } else {
        for (int it = 0; it < nk; ++it) {
            const int k0 = it * BK;
            // ---- stage A tile: 128 rows x 32 k, 2 threads/row, 16 elems ----
            {
                const int r = tid >> 1, seg = tid & 1;
                const size_t goff = aBase + (size_t)(m0 + r) * lda + k0 + seg * 16;
                unsigned short* dst = &As[r * LDSP + seg * 16];
                if constexpr (AF32) {
                    const float4* src = (const float4*)((const float*)Ap + goff);
#pragma unroll
                    for (int i = 0; i < 4; ++i) {
                        float4 v = src[i];
                        ((unsigned int*)dst)[2 * i + 0] =
                            (unsigned int)f2bf(v.x) | ((unsigned int)f2bf(v.y) << 16);
                        ((unsigned int*)dst)[2 * i + 1] =
                            (unsigned int)f2bf(v.z) | ((unsigned int)f2bf(v.w) << 16);
                    }
                } else {
                    const uint4* src = (const uint4*)((const unsigned short*)Ap + goff);
                    ((uint4*)dst)[0] = src[0];
                    ((uint4*)dst)[1] = src[1];
                }
            }
            // ---- stage B tile into Bs[n][k] (B^T, k contiguous) ------------
            if constexpr (BTRANS) {
                const int r = tid >> 1, seg = tid & 1;
                const size_t goff = bBase + (size_t)(n0 + r) * ldb + k0 + seg * 16;
                unsigned short* dst = &Bs[r * LDSP + seg * 16];
                if constexpr (BF32) {
                    const float4* src = (const float4*)((const float*)Bp + goff);
#pragma unroll
                    for (int i = 0; i < 4; ++i) {
                        float4 v = src[i];
                        ((unsigned int*)dst)[2 * i + 0] =
                            (unsigned int)f2bf(v.x) | ((unsigned int)f2bf(v.y) << 16);
                        ((unsigned int*)dst)[2 * i + 1] =
                            (unsigned int)f2bf(v.z) | ((unsigned int)f2bf(v.w) << 16);
                    }
                } else {
                    const uint4* src = (const uint4*)((const unsigned short*)Bp + goff);
                    ((uint4*)dst)[0] = src[0];
                    ((uint4*)dst)[1] = src[1];
                }
            } else {
                const int k    = tid & 31;
                const int nseg = tid >> 5;
                const size_t goff = bBase + (size_t)(k0 + k) * ldb + n0 + nseg * 16;
                if constexpr (BF32) {
                    const float4* src = (const float4*)((const float*)Bp + goff);
#pragma unroll
                    for (int i = 0; i < 4; ++i) {
                        float4 v = src[i];
                        const int nn = nseg * 16 + i * 4;
                        Bs[(nn + 0) * LDSP + k] = f2bf(v.x);
                        Bs[(nn + 1) * LDSP + k] = f2bf(v.y);
                        Bs[(nn + 2) * LDSP + k] = f2bf(v.z);
                        Bs[(nn + 3) * LDSP + k] = f2bf(v.w);
                    }
                } else {
                    const unsigned short* src = (const unsigned short*)Bp + goff;
#pragma unroll
                    for (int i = 0; i < 16; ++i)
                        Bs[(nseg * 16 + i) * LDSP + k] = src[i];
                }
            }
            __syncthreads();

            FragU aF[2], bF[4];
#pragma unroll
            for (int mi = 0; mi < 2; ++mi) {
                const int m = wm * 32 + mi * 16 + l16;
                aF[mi].u[0] = *(const uint4*)&As[m * LDSP + half * 8];
                aF[mi].u[1] = *(const uint4*)&As[m * LDSP + 16 + half * 8];
            }
#pragma unroll
            for (int ni = 0; ni < 4; ++ni) {
                const int n = wn * 64 + ni * 16 + l16;
                bF[ni].u[0] = *(const uint4*)&Bs[n * LDSP + half * 8];
                bF[ni].u[1] = *(const uint4*)&Bs[n * LDSP + 16 + half * 8];
            }
#pragma unroll
            for (int mi = 0; mi < 2; ++mi)
#pragma unroll
                for (int ni = 0; ni < 4; ++ni)
                    acc[mi][ni] = __builtin_amdgcn_wmma_f32_16x16x32_bf16(
                        false, aF[mi].v, false, bF[ni].v,
                        (short)0, acc[mi][ni], false, false);
            __syncthreads();
        }
    }

    // ---- store: C/D layout: VGPR r -> row r (lanes 0-15) / r+8 (16-31) ----
#pragma unroll
    for (int mi = 0; mi < 2; ++mi)
#pragma unroll
        for (int ni = 0; ni < 4; ++ni)
#pragma unroll
            for (int r = 0; r < 8; ++r) {
                const int m = m0 + wm * 32 + mi * 16 + half * 8 + r;
                const int n = n0 + wn * 64 + ni * 16 + l16;
                const float val = acc[mi][ni][r] * alpha;
                if constexpr (CBF16)
                    ((unsigned short*)Cp)[cBase + (size_t)m * ldc + n] = f2bf(val);
                else
                    ((float*)Cp)[cBase + (size_t)m * ldc + n] = val;
            }
}

// ---------------------------------------------------------------------------
// RoPE: read fp32 projected Q/K row (b,s), apply rotary embedding, emit bf16.
// ---------------------------------------------------------------------------
__global__ __launch_bounds__(256) void rope_to_bf16(
    const float* __restrict__ x, unsigned short* __restrict__ out,
    const int* __restrict__ pos_ids, int width)
{
    const int row = blockIdx.x;               // b*SEQ + s
    const float pos = (float)pos_ids[row];
    const size_t base = (size_t)row * width;
    for (int idx = threadIdx.x; idx < width; idx += blockDim.x) {
        const int d  = idx & (HD - 1);
        const int hd = d & (HD / 2 - 1);
        const float inv = __powf(10000.0f, -(float)(2 * hd) * (1.0f / (float)HD));
        const float ang = pos * inv;
        float sn, cs;
        __sincosf(ang, &sn, &cs);
        const float x0 = x[base + idx];
        const float other = (d < HD / 2) ? -x[base + idx + HD / 2]
                                         :  x[base + idx - HD / 2];
        out[base + idx] = f2bf(x0 * cs + other * sn);
    }
}

// ---------------------------------------------------------------------------
// Row softmax (fp32, in place) with additive mask. One block per row.
// ---------------------------------------------------------------------------
__global__ __launch_bounds__(256) void softmax_rows(
    float* __restrict__ attn, const float* __restrict__ mask)
{
    __shared__ float red[256];
    const int rid = blockIdx.x;
    const int q   = rid % SEQ;
    const int b   = rid / (NH * SEQ);
    float* row = attn + (size_t)rid * SEQ;
    const float* mrow = mask + ((size_t)b * SEQ + q) * SEQ;

    const int tid = threadIdx.x;
    float v[SEQ / 256];
    float mx = -INFINITY;
#pragma unroll
    for (int i = 0; i < SEQ / 256; ++i) {
        const int idx = tid + i * 256;
        v[i] = row[idx] + mrow[idx];
        mx = fmaxf(mx, v[i]);
    }
    red[tid] = mx;
    __syncthreads();
    for (int s = 128; s > 0; s >>= 1) {
        if (tid < s) red[tid] = fmaxf(red[tid], red[tid + s]);
        __syncthreads();
    }
    mx = red[0];
    __syncthreads();

    float sum = 0.f;
#pragma unroll
    for (int i = 0; i < SEQ / 256; ++i) {
        v[i] = __expf(v[i] - mx);
        sum += v[i];
    }
    red[tid] = sum;
    __syncthreads();
    for (int s = 128; s > 0; s >>= 1) {
        if (tid < s) red[tid] += red[tid + s];
        __syncthreads();
    }
    const float inv = 1.0f / red[0];
#pragma unroll
    for (int i = 0; i < SEQ / 256; ++i)
        row[tid + i * 256] = v[i] * inv;
}

// ---------------------------------------------------------------------------
// Launch
// ---------------------------------------------------------------------------
extern "C" void kernel_launch(void* const* d_in, const int* in_sizes, int n_in,
                              void* d_out, int out_size, void* d_ws, size_t ws_size,
                              hipStream_t stream)
{
    const float* q    = (const float*)d_in[0];
    const float* k    = (const float*)d_in[1];
    const float* v    = (const float*)d_in[2];
    const float* mask = (const float*)d_in[3];
    const int*   pos  = (const int*)  d_in[4];
    const float* Wq   = (const float*)d_in[5];
    const float* Wk   = (const float*)d_in[6];
    const float* Wv   = (const float*)d_in[7];
    const float* Wo   = (const float*)d_in[8];

    const long long BS = (long long)BATCH * SEQ;           // 4096 rows
    float* out  = (float*)d_out;                            // (B,S,HID)
    float* attn = out + BS * HID;                           // (B,NH,S,S)

    // workspace carve-up
    char* ws = (char*)d_ws;
    float*          Qf  = (float*)ws;            ws += BS * HID * sizeof(float);      // 33.5 MB
    float*          Kf  = (float*)ws;            ws += BS * (NKV*HD) * sizeof(float); //  8.4 MB
    unsigned short* Qr  = (unsigned short*)ws;   ws += BS * HID * 2;                  // 16.8 MB
    unsigned short* Kr  = (unsigned short*)ws;   ws += BS * (NKV*HD) * 2;             //  4.2 MB
    unsigned short* Vh  = (unsigned short*)ws;   ws += BS * (NKV*HD) * 2;             //  4.2 MB
    unsigned short* ctx = (unsigned short*)ws;                                        // 16.8 MB

    const dim3 blk(256);

    // 1) Q/K/V projections (A fp32, B fp32 KxN, WMMA bf16)
    gemm_wmma_bf16<true, true, false, false><<<dim3(HID/BN, BS/BM, 1), blk, 0, stream>>>(
        q, HID, Wq, NH*HD, Qf, NH*HD, HID, 1.0f, 0,0, 0,0,1, 0,0, 1);
    gemm_wmma_bf16<true, true, false, false><<<dim3((NKV*HD)/BN, BS/BM, 1), blk, 0, stream>>>(
        k, HID, Wk, NKV*HD, Kf, NKV*HD, HID, 1.0f, 0,0, 0,0,1, 0,0, 1);
    gemm_wmma_bf16<true, true, false, true><<<dim3((NKV*HD)/BN, BS/BM, 1), blk, 0, stream>>>(
        v, HID, Wv, NKV*HD, Vh, NKV*HD, HID, 1.0f, 0,0, 0,0,1, 0,0, 1);

    // 2) RoPE on Q and K -> bf16
    rope_to_bf16<<<dim3(BATCH*SEQ), blk, 0, stream>>>(Qf, Qr, pos, NH*HD);
    rope_to_bf16<<<dim3(BATCH*SEQ), blk, 0, stream>>>(Kf, Kr, pos, NKV*HD);

    // 3) scores = scale * Q @ K^T per (b,h); TDM + double-buffered staging
    gemm_wmma_bf16<false, false, true, false><<<dim3(SEQ/BN, SEQ/BM, BATCH*NH), blk, 0, stream>>>(
        Qr, NH*HD, Kr, NKV*HD, attn, SEQ, HD, SCALE,
        (long long)SEQ*NH*HD, HD,
        (long long)SEQ*NKV*HD, HD, GROUPS,
        (long long)NH*SEQ*SEQ, (long long)SEQ*SEQ, NH);

    // 4) softmax with additive mask (in place, fp32)
    softmax_rows<<<dim3(BATCH*NH*SEQ), blk, 0, stream>>>(attn, mask);

    // 5) ctx = attn @ V  (A fp32 probs, B bf16, C bf16 into (B,S,NH*HD))
    gemm_wmma_bf16<true, false, false, true><<<dim3(HD/BN, SEQ/BM, BATCH*NH), blk, 0, stream>>>(
        attn, SEQ, Vh, NKV*HD, ctx, NH*HD, SEQ, 1.0f,
        (long long)NH*SEQ*SEQ, (long long)SEQ*SEQ,
        (long long)SEQ*NKV*HD, HD, GROUPS,
        (long long)SEQ*NH*HD, HD, NH);

    // 6) out = ctx @ Wo (fp32 result)
    gemm_wmma_bf16<false, true, false, false><<<dim3(HID/BN, BS/BM, 1), blk, 0, stream>>>(
        ctx, NH*HD, Wo, HID, out, HID, NH*HD, 1.0f, 0,0, 0,0,1, 0,0, 1);
}
